// Aggre_user_27814208209715
// MI455X (gfx1250) — matrix operand order repaired and checked by reading, not compiled, not run on hardware
//
#include <hip/hip_runtime.h>
#include <hip/hip_bf16.h>

typedef __bf16 bf16_t;
typedef __attribute__((ext_vector_type(16))) __bf16 v16bf;
typedef __attribute__((ext_vector_type(8)))  float  v8f;

#define DEV __device__ __forceinline__

// ---------------------------------------------------------------------------
// WMMA operand builders (CDNA5 wave32 layouts, CDNA5 ISA §7.12.2)
//
// A (16x32 bf16, M x K): lane L (half h = L>>4) holds row M = L&15,
//   elements 0..7  -> K = kb + 8h + {0..7}
//   elements 8..15 -> K = kb + 16 + 8h + {0..7}
// B (32x16 bf16, K x N): lane L holds column N = L&15,
//   elements 0..7  -> K = kb + 8h + {0..7}
//   elements 8..15 -> K = kb + 16 + 8h + {0..7}
// C/D (16x16 f32): lane L holds column N = L&15, element v -> row M = v + 8h
// ---------------------------------------------------------------------------

DEV v16bf a_chunk_f32(const float* __restrict__ row, int kb, int half) {
  v16bf a;
  const float* p0 = row + kb + half * 8;
  const float* p1 = p0 + 16;
#pragma unroll
  for (int i = 0; i < 8; ++i) { a[i] = (__bf16)p0[i]; a[8 + i] = (__bf16)p1[i]; }
  return a;
}

DEV v16bf a_chunk_bf16(const bf16_t* __restrict__ row, int kb, int half) {
  v16bf a;
  const bf16_t* p0 = row + kb + half * 8;
  const bf16_t* p1 = p0 + 16;
#pragma unroll
  for (int i = 0; i < 8; ++i) { a[i] = p0[i]; a[8 + i] = p1[i]; }
  return a;
}

DEV v16bf b_chunk(const bf16_t* __restrict__ w, int ldk, int n, int kb, int half) {
  v16bf b;
  const bf16_t* p0 = w + n * ldk + kb + half * 8;
  const bf16_t* p1 = p0 + 16;
#pragma unroll
  for (int i = 0; i < 8; ++i) { b[i] = p0[i]; b[8 + i] = p1[i]; }
  return b;
}

DEV v8f wmma_bf16(v16bf a, v16bf b, v8f c) {
  return __builtin_amdgcn_wmma_f32_16x16x32_bf16(false, a, false, b, (short)0, c,
                                                 false, false);
}

DEV void zero_acc(v8f* acc) {
#pragma unroll
  for (int t = 0; t < 4; ++t)
#pragma unroll
    for (int i = 0; i < 8; ++i) acc[t][i] = 0.0f;
}

#define LDS_FENCE() asm volatile("s_wait_dscnt 0x0" ::: "memory")

// ---------------------------------------------------------------------------
// Kernel 1: per-edge MLP.  Computes f[E,64] (fp32) and attention logits a[E].
// One 16-edge tile per wave; 40 v_wmma_f32_16x16x32_bf16 per tile.
// ---------------------------------------------------------------------------
__global__ __launch_bounds__(256) void edge_kernel(
    const float* __restrict__ user_table, const float* __restrict__ rating_table,
    const float* __restrict__ item_table,
    const float* __restrict__ ln1_w, const float* __restrict__ ln1_b,
    const float* __restrict__ att1_w, const float* __restrict__ att1_b,
    const float* __restrict__ att2_w, const float* __restrict__ att2_b,
    const float* __restrict__ att3_w, const float* __restrict__ att3_b,
    const int* __restrict__ item_idx, const int* __restrict__ user_idx,
    const int* __restrict__ rating_idx, const int* __restrict__ seg,
    float* __restrict__ f_out, float* __restrict__ a_out, int E) {
  __shared__ bf16_t s_w1[64 * 128];   // ln1_w  (bf16)
  __shared__ bf16_t s_wa1[64 * 128];  // att1_w (bf16)
  __shared__ bf16_t s_wa2[64 * 64];   // att2_w (bf16)
  __shared__ float  s_b1[64], s_ba1[64], s_ba2[64], s_w3[64];
  __shared__ float  s_b3;
  __shared__ bf16_t s_stage[8][16 * 64];  // per-wave transpose staging

  const int tid = threadIdx.x;
  for (int i = tid; i < 64 * 128; i += 256) {
    s_w1[i]  = (__bf16)ln1_w[i];
    s_wa1[i] = (__bf16)att1_w[i];
  }
  for (int i = tid; i < 64 * 64; i += 256) s_wa2[i] = (__bf16)att2_w[i];
  if (tid < 64) {
    s_b1[tid]  = ln1_b[tid];
    s_ba1[tid] = att1_b[tid];
    s_ba2[tid] = att2_b[tid];
    s_w3[tid]  = att3_w[tid];
  }
  if (tid == 0) s_b3 = att3_b[0];
  __syncthreads();

  const int lane = tid & 31;
  const int wave = tid >> 5;
  const int half = lane >> 4;
  const int nl   = lane & 15;
  bf16_t* stage = s_stage[wave];

  const int ntiles = E >> 4;  // E is a multiple of 16 (409600)
  for (int tile = blockIdx.x * 8 + wave; tile < ntiles; tile += gridDim.x * 8) {
    const int e0 = tile * 16;
    int er = e0 + nl;
    if (er >= E) er = E - 1;  // safety clamp (no-op when E % 16 == 0)
    const float* urow = user_table   + (size_t)user_idx[er] * 64;
    const float* rrow = rating_table + (size_t)rating_idx[er] * 64;
    const float* irow = item_table   + (size_t)item_idx[seg[er]] * 64;

    // ---- f = relu([w_user, w_rating] @ ln1_w^T + ln1_b) ----
    v8f acc[4];
    zero_acc(acc);
#pragma unroll
    for (int kc = 0; kc < 4; ++kc) {
      const int kb = kc * 32;
      v16bf a = (kb < 64) ? a_chunk_f32(urow, kb, half)
                          : a_chunk_f32(rrow, kb - 64, half);
#pragma unroll
      for (int nt = 0; nt < 4; ++nt)
        acc[nt] = wmma_bf16(a, b_chunk(s_w1, 128, nl + nt * 16, kb, half), acc[nt]);
    }
    LDS_FENCE();
#pragma unroll
    for (int nt = 0; nt < 4; ++nt) {
      const int n = nl + nt * 16;
      const float bias = s_b1[n];
#pragma unroll
      for (int v = 0; v < 8; ++v) {
        const int m = v + half * 8;
        const float x = fmaxf(acc[nt][v] + bias, 0.0f);
        if (e0 + m < E) f_out[(size_t)(e0 + m) * 64 + n] = x;
        stage[m * 64 + n] = (__bf16)x;
      }
    }
    LDS_FENCE();

    // ---- h = relu([f, w_item] @ att1_w^T + att1_b) ----
    zero_acc(acc);
#pragma unroll
    for (int kc = 0; kc < 4; ++kc) {
      const int kb = kc * 32;
      v16bf a = (kb < 64) ? a_chunk_bf16(stage + nl * 64, kb, half)
                          : a_chunk_f32(irow, kb - 64, half);
#pragma unroll
      for (int nt = 0; nt < 4; ++nt)
        acc[nt] = wmma_bf16(a, b_chunk(s_wa1, 128, nl + nt * 16, kb, half), acc[nt]);
    }
    LDS_FENCE();
#pragma unroll
    for (int nt = 0; nt < 4; ++nt) {
      const int n = nl + nt * 16;
      const float bias = s_ba1[n];
#pragma unroll
      for (int v = 0; v < 8; ++v) {
        const int m = v + half * 8;
        stage[m * 64 + n] = (__bf16)fmaxf(acc[nt][v] + bias, 0.0f);
      }
    }
    LDS_FENCE();

    // ---- h = relu(h @ att2_w^T + att2_b) ----
    zero_acc(acc);
#pragma unroll
    for (int kc = 0; kc < 2; ++kc) {
      const int kb = kc * 32;
      v16bf a = a_chunk_bf16(stage + nl * 64, kb, half);
#pragma unroll
      for (int nt = 0; nt < 4; ++nt)
        acc[nt] = wmma_bf16(a, b_chunk(s_wa2, 64, nl + nt * 16, kb, half), acc[nt]);
    }

    // ---- a = h @ att3_w^T + att3_b  (64 -> 1) ----
    float part[8];
#pragma unroll
    for (int v = 0; v < 8; ++v) part[v] = 0.0f;
#pragma unroll
    for (int nt = 0; nt < 4; ++nt) {
      const int n = nl + nt * 16;
      const float bias = s_ba2[n];
      const float w3 = s_w3[n];
#pragma unroll
      for (int v = 0; v < 8; ++v)
        part[v] += fmaxf(acc[nt][v] + bias, 0.0f) * w3;
    }
    // reduce over the 16 lanes of each half (N dimension)
#pragma unroll
    for (int msk = 1; msk < 16; msk <<= 1)
#pragma unroll
      for (int v = 0; v < 8; ++v) part[v] += __shfl_xor(part[v], msk, 32);
    if (nl < 8) {
      float av = part[0];
#pragma unroll
      for (int v = 1; v < 8; ++v)
        if (nl == v) av = part[v];
      const int m = half * 8 + nl;
      if (e0 + m < E) a_out[e0 + m] = av + s_b3;
    }
  }
}

// ---------------------------------------------------------------------------
// Kernel 2: segment boundaries by binary search (segment_ids sorted).
// ---------------------------------------------------------------------------
DEV int lower_bound_seg(const int* __restrict__ seg, int E, int key) {
  int lo = 0, hi = E;
  while (lo < hi) {
    const int mid = (lo + hi) >> 1;
    if (seg[mid] < key) lo = mid + 1; else hi = mid;
  }
  return lo;
}

__global__ __launch_bounds__(256) void bounds_kernel(const int* __restrict__ seg,
                                                     int* __restrict__ starts,
                                                     int* __restrict__ ends,
                                                     int E, int B) {
  const int s = blockIdx.x * blockDim.x + threadIdx.x;
  if (s >= B) return;
  starts[s] = lower_bound_seg(seg, E, s);
  ends[s]   = lower_bound_seg(seg, E, s + 1);
}

// ---------------------------------------------------------------------------
// Kernel 3: segment softmax + attention-weighted sum.  One wave per segment.
// z[s,:] = sum_e (exp(a[e]-max)/max(den,1e-9)) * f[e,:]
// ---------------------------------------------------------------------------
__global__ __launch_bounds__(256) void seg_kernel(const float* __restrict__ f,
                                                  const float* __restrict__ a,
                                                  const int* __restrict__ starts,
                                                  const int* __restrict__ ends,
                                                  float* __restrict__ z, int B) {
  const int gwave = (blockIdx.x * blockDim.x + threadIdx.x) >> 5;
  const int lane  = threadIdx.x & 31;
  if (gwave >= B) return;
  const int s0 = starts[gwave];
  const int s1 = ends[gwave];

  float m = -INFINITY;
  for (int e = s0 + lane; e < s1; e += 32) m = fmaxf(m, a[e]);
#pragma unroll
  for (int msk = 16; msk >= 1; msk >>= 1) m = fmaxf(m, __shfl_xor(m, msk, 32));

  float den = 0.0f;
  for (int e = s0 + lane; e < s1; e += 32) den += __expf(a[e] - m);
#pragma unroll
  for (int msk = 16; msk >= 1; msk >>= 1) den += __shfl_xor(den, msk, 32);
  const float inv = 1.0f / fmaxf(den, 1e-9f);

  float acc0 = 0.0f, acc1 = 0.0f;
  for (int e = s0; e < s1; ++e) {
    const float mu = __expf(a[e] - m) * inv;
    const float* fr = f + (size_t)e * 64;
    acc0 += mu * fr[lane];
    acc1 += mu * fr[32 + lane];
  }
  z[(size_t)gwave * 64 + lane]      = acc0;
  z[(size_t)gwave * 64 + 32 + lane] = acc1;
}

// ---------------------------------------------------------------------------
// Kernel 4: per-node MLP.  z2 = relu(z @ ln2_w^T + b2);
// out = relu([node_item, z2] @ ln3_w^T + b3).  16-node tile per wave.
// ---------------------------------------------------------------------------
__global__ __launch_bounds__(256) void node_kernel(
    const float* __restrict__ item_table,
    const float* __restrict__ ln2_w, const float* __restrict__ ln2_b,
    const float* __restrict__ ln3_w, const float* __restrict__ ln3_b,
    const int* __restrict__ item_idx, const float* __restrict__ z_in,
    float* __restrict__ out, int B) {
  __shared__ bf16_t s_w2[64 * 64];
  __shared__ bf16_t s_w3[64 * 128];
  __shared__ float  s_b2[64], s_b3[64];
  __shared__ bf16_t s_stage[8][16 * 64];

  const int tid = threadIdx.x;
  for (int i = tid; i < 64 * 128; i += 256) s_w3[i] = (__bf16)ln3_w[i];
  for (int i = tid; i < 64 * 64; i += 256)  s_w2[i] = (__bf16)ln2_w[i];
  if (tid < 64) { s_b2[tid] = ln2_b[tid]; s_b3[tid] = ln3_b[tid]; }
  __syncthreads();

  const int lane = tid & 31;
  const int wave = tid >> 5;
  const int half = lane >> 4;
  const int nl   = lane & 15;
  bf16_t* stage = s_stage[wave];

  const int ntiles = B >> 4;  // B is a multiple of 16 (8192)
  for (int tile = blockIdx.x * 8 + wave; tile < ntiles; tile += gridDim.x * 8) {
    const int n0 = tile * 16;
    int nr = n0 + nl;
    if (nr >= B) nr = B - 1;
    const float* zrow = z_in + (size_t)nr * 64;
    const float* irow = item_table + (size_t)item_idx[nr] * 64;

    // ---- z2 = relu(z @ ln2_w^T + b2) ----
    v8f acc[4];
    zero_acc(acc);
#pragma unroll
    for (int kc = 0; kc < 2; ++kc) {
      const int kb = kc * 32;
      v16bf a = a_chunk_f32(zrow, kb, half);
#pragma unroll
      for (int nt = 0; nt < 4; ++nt)
        acc[nt] = wmma_bf16(a, b_chunk(s_w2, 64, nl + nt * 16, kb, half), acc[nt]);
    }
    LDS_FENCE();
#pragma unroll
    for (int nt = 0; nt < 4; ++nt) {
      const int n = nl + nt * 16;
      const float bias = s_b2[n];
#pragma unroll
      for (int v = 0; v < 8; ++v) {
        const int m = v + half * 8;
        stage[m * 64 + n] = (__bf16)fmaxf(acc[nt][v] + bias, 0.0f);
      }
    }
    LDS_FENCE();

    // ---- out = relu([node_item, z2] @ ln3_w^T + b3) ----
    zero_acc(acc);
#pragma unroll
    for (int kc = 0; kc < 4; ++kc) {
      const int kb = kc * 32;
      v16bf a = (kb < 64) ? a_chunk_f32(irow, kb, half)
                          : a_chunk_bf16(stage + nl * 64, kb - 64, half);
#pragma unroll
      for (int nt = 0; nt < 4; ++nt)
        acc[nt] = wmma_bf16(a, b_chunk(s_w3, 128, nl + nt * 16, kb, half), acc[nt]);
    }
#pragma unroll
    for (int nt = 0; nt < 4; ++nt) {
      const int n = nl + nt * 16;
      const float bias = s_b3[n];
#pragma unroll
      for (int v = 0; v < 8; ++v) {
        const int m = v + half * 8;
        if (n0 + m < B)
          out[(size_t)(n0 + m) * 64 + n] = fmaxf(acc[nt][v] + bias, 0.0f);
      }
    }
  }
}

// ---------------------------------------------------------------------------
extern "C" void kernel_launch(void* const* d_in, const int* in_sizes, int n_in,
                              void* d_out, int out_size, void* d_ws, size_t ws_size,
                              hipStream_t stream) {
  const float* user_table   = (const float*)d_in[0];
  const float* rating_table = (const float*)d_in[1];
  const float* item_table   = (const float*)d_in[2];
  const float* ln1_w  = (const float*)d_in[3];
  const float* ln1_b  = (const float*)d_in[4];
  const float* ln2_w  = (const float*)d_in[5];
  const float* ln2_b  = (const float*)d_in[6];
  const float* ln3_w  = (const float*)d_in[7];
  const float* ln3_b  = (const float*)d_in[8];
  const float* att1_w = (const float*)d_in[9];
  const float* att1_b = (const float*)d_in[10];
  const float* att2_w = (const float*)d_in[11];
  const float* att2_b = (const float*)d_in[12];
  const float* att3_w = (const float*)d_in[13];
  const float* att3_b = (const float*)d_in[14];
  const int* item_idx   = (const int*)d_in[15];
  const int* user_idx   = (const int*)d_in[16];
  const int* rating_idx = (const int*)d_in[17];
  const int* seg        = (const int*)d_in[18];

  const int Bn = in_sizes[15];  // 8192
  const int E  = in_sizes[16];  // 409600

  // workspace layout
  char* ws = (char*)d_ws;
  float* f_buf = (float*)ws;  ws += (size_t)E * 64 * sizeof(float);
  float* a_buf = (float*)ws;  ws += (size_t)E * sizeof(float);
  int* starts  = (int*)ws;    ws += (size_t)Bn * sizeof(int);
  int* ends    = (int*)ws;    ws += (size_t)Bn * sizeof(int);
  float* z_buf = (float*)ws;

  edge_kernel<<<512, 256, 0, stream>>>(
      user_table, rating_table, item_table, ln1_w, ln1_b, att1_w, att1_b,
      att2_w, att2_b, att3_w, att3_b, item_idx, user_idx, rating_idx, seg,
      f_buf, a_buf, E);

  bounds_kernel<<<(Bn + 255) / 256, 256, 0, stream>>>(seg, starts, ends, E, Bn);

  seg_kernel<<<(Bn * 32 + 255) / 256, 256, 0, stream>>>(f_buf, a_buf, starts,
                                                        ends, z_buf, Bn);

  node_kernel<<<64, 256, 0, stream>>>(item_table, ln2_w, ln2_b, ln3_w, ln3_b,
                                      item_idx, z_buf, (float*)d_out, Bn);
}
